// tPatchGNN_43301860278570
// MI455X (gfx1250) — matrix operand order repaired and checked by reading, not compiled, not run on hardware
//
#include <hip/hip_runtime.h>
#include <hip/hip_bf16.h>
#include <math.h>

#define DEV __device__ __forceinline__

typedef __attribute__((ext_vector_type(8)))  _Float16 v8h;
typedef __attribute__((ext_vector_type(16))) _Float16 v16h;
typedef __attribute__((ext_vector_type(8)))  float    v8f;

// ---------------- model constants ----------------
#define B_    8
#define N_    32
#define M_    16
#define LX    32
#define D_    64
#define INDIM 17
#define TT_   63
#define P_    (B_*N_*M_)      // 4096
#define NTIL3 67              // ceil(1071/16)
#define W3N   1072            // padded N for w3
#define W3K   64              // padded K for w3

DEV int iMin(int a, int b){ return a < b ? a : b; }

// ---------------- WMMA fragment helpers (CDNA5 16x16x32 f16) ----------------
// A (16xK slice of 32): lanes 0-15 row=lane, elems 0-15 => K k0+0..15
//                       lanes 16-31 row=lane-16, elems 0-15 => K k0+16..31.
DEV v16h load_a_frag(const _Float16* src, int ld, int lane, int k0){
  int row = lane & 15;
  int kh  = (lane & 16);                // 0 or 16
  const _Float16* p = src + row*ld + k0 + kh;
  v8h lo = *(const v8h*)p;
  v8h hi = *(const v8h*)(p + 8);
  v16h a;
#pragma unroll
  for(int e=0;e<8;++e){ a[e]=lo[e]; a[e+8]=hi[e]; }
  return a;
}
// B[k][n] = W[n][k] with W row-major (N,K). lanes 0-15 col=n0+lane, K k0..k0+15; lanes 16-31 K k0+16..31.
DEV v16h load_b_frag(const _Float16* W, int ld, int lane, int n0, int k0){
  int col = n0 + (lane & 15);
  int kb  = k0 + (lane & 16);           // +0 or +16
  return *(const v16h*)(W + (size_t)col*ld + kb);
}
DEV v8f wmma_f16(v16h a, v16h b, v8f c){
  return __builtin_amdgcn_wmma_f32_16x16x32_f16(false, a, false, b, (short)0, c, false, false);
}

// OUT[16][Nout](fp32, stride ldOut, col offset co) = A16[16][K] @ W^T + bias
DEV void gemm16(const _Float16* A, int K, const _Float16* W, const float* bias,
                float* OUT, int ldOut, int co, int Nout){
  int wave = threadIdx.x >> 5, lane = threadIdx.x & 31;
  int nw = blockDim.x >> 5;
  int ntiles = Nout >> 4, ksteps = K >> 5;
  for(int nt = wave; nt < ntiles; nt += nw){
    v8f acc = {};
    for(int k = 0; k < ksteps; ++k){
      v16h a = load_a_frag(A, K, lane, k*32);
      v16h b = load_b_frag(W, K, lane, nt*16, k*32);
      acc = wmma_f16(a, b, acc);
    }
    int col = nt*16 + (lane & 15);
    int rb  = (lane & 16) >> 1;
    float bb = bias ? bias[col] : 0.f;
#pragma unroll
    for(int r = 0; r < 8; ++r) OUT[(rb + r)*ldOut + co + col] = acc[r] + bb;
  }
}

// three 64-wide projections (q,k,v) from one A, into OUT[16][192]
DEV void gemm_qkv(const _Float16* A,
                  const _Float16* qw, const float* qb,
                  const _Float16* kw, const float* kb,
                  const _Float16* vw, const float* vb, float* OUT){
  int wave = threadIdx.x >> 5, lane = threadIdx.x & 31;
  int nw = blockDim.x >> 5;
  for(int u = wave; u < 12; u += nw){
    int s = u >> 2, nt = u & 3;
    const _Float16* W = (s==0) ? qw : (s==1) ? kw : vw;
    const float*    Bv= (s==0) ? qb : (s==1) ? kb : vb;
    v8f acc = {};
    for(int k = 0; k < 2; ++k){
      v16h a = load_a_frag(A, 64, lane, k*32);
      v16h b = load_b_frag(W, 64, lane, nt*16, k*32);
      acc = wmma_f16(a, b, acc);
    }
    int col = nt*16 + (lane & 15);
    int rb  = (lane & 16) >> 1;
    float bb = Bv[col];
#pragma unroll
    for(int r = 0; r < 8; ++r) OUT[(rb + r)*192 + s*64 + col] = acc[r] + bb;
  }
}

// 4-head 16x16 attention, VALU. qkv stride 192 (q @0, k @64, v @128). out stride 64.
DEV void attention16(const float* qkv, float* out, float scale,
                     const float* tb, const float* pmr){
  int tid = threadIdx.x;
  if(tid < 64){
    int h = tid >> 4, row = tid & 15;
    const float* q = qkv + row*192 + h*16;
    float sc[16]; float mx = -3.0e38f;
#pragma unroll
    for(int c = 0; c < 16; ++c){
      const float* kk = qkv + c*192 + 64 + h*16;
      float s = 0.f;
#pragma unroll
      for(int d = 0; d < 16; ++d) s += q[d]*kk[d];
      s *= scale;
      if(tb)  s += tb[row*16 + c];
      if(pmr && pmr[c] == 0.f) s = -1.0e9f;
      sc[c] = s; mx = fmaxf(mx, s);
    }
    float sum = 0.f;
#pragma unroll
    for(int c = 0; c < 16; ++c){ sc[c] = __expf(sc[c] - mx); sum += sc[c]; }
    float inv = 1.f / sum;
#pragma unroll
    for(int d = 0; d < 16; ++d){
      float o = 0.f;
#pragma unroll
      for(int c = 0; c < 16; ++c) o += sc[c]*qkv[c*192 + 128 + h*16 + d];
      out[row*64 + h*16 + d] = o*inv;
    }
  }
}

// S = LayerNorm(S + R)
DEV void ln_add(float* S, const float* R, const float* g, const float* b){
  int tid = threadIdx.x;
  if(tid < 16){
    float* row = S + tid*64; const float* rr = R + tid*64;
    float mu = 0.f;
    for(int d = 0; d < 64; ++d){ row[d] += rr[d]; mu += row[d]; }
    mu *= (1.f/64.f);
    float var = 0.f;
    for(int d = 0; d < 64; ++d){ float t = row[d]-mu; var += t*t; }
    var *= (1.f/64.f);
    float inv = rsqrtf(var + 1e-5f);
    for(int d = 0; d < 64; ++d) row[d] = (row[d]-mu)*inv*g[d] + b[d];
  }
}

DEV float pos_enc(int m, int d){
  float div = __expf(-0.14390932f * (float)(d & ~1));   // ln(10000)/64 * 2i
  float ang = (float)m * div;
  return (d & 1) ? __cosf(ang) : __sinf(ang);
}

// ---------------- weight conversion ----------------
__global__ void cvt_f16(const float* __restrict__ src, _Float16* __restrict__ dst, int n){
  int i = blockIdx.x*blockDim.x + threadIdx.x;
  if(i < n) dst[i] = (_Float16)src[i];
}
__global__ void cvt_w3(const float* __restrict__ src, _Float16* __restrict__ dst){
  int i = blockIdx.x*blockDim.x + threadIdx.x;
  if(i < W3N*W3K){
    int n = i >> 6, k = i & 63;
    dst[i] = (n < 1071 && k < TT_) ? (_Float16)src[n*TT_ + k] : (_Float16)0.f;
  }
}

// ---------------- K1: fused TTCN (one block per patch p) ----------------
__global__ __launch_bounds__(256) void ttcn_kernel(
    const float* __restrict__ x, const int* __restrict__ mask,
    const float* __restrict__ w1, const float* __restrict__ b1,
    const float* __restrict__ w2, const float* __restrict__ b2,
    const _Float16* __restrict__ w3h, const float* __restrict__ b3,
    const float* __restrict__ tbias,
    float* __restrict__ xp, float* __restrict__ pmOut)
{
  __shared__ float sx[LX*INDIM];
  __shared__ float sm[LX];
  __shared__ float sh1[LX*64];
  __shared__ __align__(32) _Float16 sh2[LX*64];
  __shared__ float sf[LX*272];           // one column-chunk of filt
  __shared__ float sht[64];
  int p = blockIdx.x, tid = threadIdx.x, wave = tid >> 5, lane = tid & 31;

  for(int i = tid; i < LX*INDIM; i += 256) sx[i] = x[(size_t)p*LX*INDIM + i];
  if(tid < LX) sm[tid] = mask[p*LX + tid] ? 1.f : 0.f;
  if(tid < 64) sht[tid] = 0.f;
  __syncthreads();

  // h1 = relu(x @ w1^T + b1)   (32x17 -> 32x63)
  for(int idx = tid; idx < LX*TT_; idx += 256){
    int l = idx / TT_, j = idx - l*TT_;
    float a = b1[j];
    for(int i = 0; i < INDIM; ++i) a += sx[l*INDIM + i]*w1[j*INDIM + i];
    sh1[l*64 + j] = fmaxf(a, 0.f);
  }
  if(tid < LX) sh1[tid*64 + 63] = 0.f;
  __syncthreads();

  // h2 = relu(h1 @ w2^T + b2)  -> f16 A-tile (K padded to 64)
  for(int idx = tid; idx < LX*TT_; idx += 256){
    int l = idx / TT_, j = idx - l*TT_;
    float a = b2[j];
    for(int k = 0; k < TT_; ++k) a += sh1[l*64 + k]*w2[j*TT_ + k];
    sh2[l*64 + j] = (_Float16)fmaxf(a, 0.f);
  }
  if(tid < LX) sh2[tid*64 + 63] = (_Float16)0.f;
  __syncthreads();

  // filt = h2 @ w3^T + b3 in 4 column chunks of 272; softmax over l; einsum into sht
  for(int chunk = 0; chunk < 4; ++chunk){
    int nt0 = chunk*17;
    int ntc = iMin(17, NTIL3 - nt0);
    for(int tile = wave; tile < 2*ntc; tile += 8){
      int mt = tile & 1, ntl = tile >> 1;
      v8f acc = {};
#pragma unroll
      for(int k = 0; k < 2; ++k){
        v16h a = load_a_frag(sh2 + mt*16*64, 64, lane, k*32);
        v16h b = load_b_frag(w3h, W3K, lane, (nt0 + ntl)*16, k*32);
        acc = wmma_f16(a, b, acc);
      }
      int gc = (nt0 + ntl)*16 + (lane & 15);
      int cc = gc - chunk*272;
      int rb = ((lane & 16) >> 1) + mt*16;
      float bb = (gc < 1071) ? b3[gc] : 0.f;
#pragma unroll
      for(int r = 0; r < 8; ++r) sf[(rb + r)*272 + cc] = acc[r] + bb;
    }
    __syncthreads();
    int ncols = iMin(272, 1071 - chunk*272);
    for(int cc = tid; cc < ncols; cc += 256){
      int gc = chunk*272 + cc;
      int t = gc / INDIM, i2 = gc - t*INDIM;
      float mx = -3.0e38f;
      for(int l = 0; l < LX; ++l){
        float v = (sm[l] != 0.f) ? sf[l*272 + cc] : -1.0e8f;
        mx = fmaxf(mx, v);
      }
      float sum = 0.f;
      for(int l = 0; l < LX; ++l){
        float v = (sm[l] != 0.f) ? sf[l*272 + cc] : -1.0e8f;
        sum += __expf(v - mx);
      }
      float acc2 = 0.f;
      for(int l = 0; l < LX; ++l){
        float v = (sm[l] != 0.f) ? sf[l*272 + cc] : -1.0e8f;
        acc2 += __expf(v - mx)*sx[l*INDIM + i2];
      }
      atomicAdd(&sht[t], acc2 / sum);
    }
    __syncthreads();
  }

  if(tid < TT_){
    float v = fmaxf(sht[tid] + tbias[tid], 0.f);
    xp[(size_t)p*64 + tid] = v;
  }
  if(tid == TT_){
    float s = 0.f;
    for(int l = 0; l < LX; ++l) s += sm[l];
    float mp = (s > 0.f) ? 1.f : 0.f;
    xp[(size_t)p*64 + 63] = mp;
    pmOut[p] = mp;
  }
}

// ---------------- K2: one transformer layer (one block per (b,n) sequence) ----------------
__global__ __launch_bounds__(256) void transformer_layer(
    const float* __restrict__ xin, float* __restrict__ xout, const float* __restrict__ pm,
    const _Float16* __restrict__ qw, const float* __restrict__ qb,
    const _Float16* __restrict__ kw, const float* __restrict__ kb,
    const _Float16* __restrict__ vw, const float* __restrict__ vb,
    const _Float16* __restrict__ ow, const float* __restrict__ ob,
    const _Float16* __restrict__ inw, const float* __restrict__ inb,
    const _Float16* __restrict__ outw, const float* __restrict__ outb,
    const float* __restrict__ ln1g, const float* __restrict__ ln1b,
    const float* __restrict__ ln2g, const float* __restrict__ ln2b,
    const _Float16* __restrict__ fw1, const float* __restrict__ fb1,
    const _Float16* __restrict__ fw2, const float* __restrict__ fb2,
    int add_input)
{
  __shared__ float S[16*64], X0[16*64], QKV[16*192], ATT[16*64], TMP[16*64];
  __shared__ float TB[16*16], PMr[16], TMP2[2*16*64];
  __shared__ __align__(32) _Float16 A16[16*64];
  __shared__ __align__(32) _Float16 HID[16*512];
  int bn = blockIdx.x, tid = threadIdx.x, wave = tid >> 5, lane = tid & 31;

  __builtin_prefetch(fw1, 0, 2);
  __builtin_prefetch(fw2, 0, 2);

  for(int i = tid; i < 1024; i += 256){ float v = xin[(size_t)bn*1024 + i]; S[i] = v; X0[i] = v; }
  if(tid < 16) PMr[tid] = pm[bn*16 + tid];
  for(int i = tid; i < 256; i += 256){
    int r = i >> 4, c = i & 15;
    float d = fabsf((float)(r - c));
    TB[i] = logf(__expf(-5.f*d) + 1e-12f);   // log(exp(-|i-j|*300/60)+1e-12)
  }
  __syncthreads();

  // ---- patch gattn ----
  for(int i = tid; i < 1024; i += 256) A16[i] = (_Float16)S[i];
  __syncthreads();
  gemm_qkv(A16, qw, qb, kw, kb, vw, vb, QKV);
  __syncthreads();
  attention16(QKV, ATT, 0.25f, TB, PMr);
  __syncthreads();
  for(int i = tid; i < 1024; i += 256) A16[i] = (_Float16)ATT[i];
  __syncthreads();
  gemm16(A16, 64, ow, ob, TMP, 64, 0, 64);
  __syncthreads();
  for(int i = tid; i < 1024; i += 256){
    int m = i >> 6, d = i & 63;
    S[i] += TMP[i]*PMr[m] + pos_enc(m, d);     // residual * pm, then + pe
  }
  __syncthreads();

  // ---- transformer MHA ----
  for(int i = tid; i < 1024; i += 256) A16[i] = (_Float16)S[i];
  __syncthreads();
  gemm16(A16, 64, inw, inb, QKV, 192, 0, 192);
  __syncthreads();
  attention16(QKV, ATT, 0.25f, nullptr, nullptr);
  __syncthreads();
  for(int i = tid; i < 1024; i += 256) A16[i] = (_Float16)ATT[i];
  __syncthreads();
  gemm16(A16, 64, outw, outb, TMP, 64, 0, 64);
  __syncthreads();
  ln_add(S, TMP, ln1g, ln1b);
  __syncthreads();

  // ---- FFN: hidden 2048 in 4 chunks of 512 (f16), acc stays in VGPRs ----
  for(int i = tid; i < 1024; i += 256) A16[i] = (_Float16)S[i];
  __syncthreads();
  v8f acc = {};
  int ntO = wave & 3, khO = wave >> 2;
  for(int chunk = 0; chunk < 4; ++chunk){
    for(int t = wave; t < 32; t += 8){
      v8f a = {};
#pragma unroll
      for(int k = 0; k < 2; ++k){
        v16h av = load_a_frag(A16, 64, lane, k*32);
        v16h bv = load_b_frag(fw1, 64, lane, chunk*512 + t*16, k*32);
        a = wmma_f16(av, bv, a);
      }
      int col = t*16 + (lane & 15);
      int gcol = chunk*512 + col;
      int rb = (lane & 16) >> 1;
      float bb = fb1[gcol];
#pragma unroll
      for(int r = 0; r < 8; ++r) HID[(rb + r)*512 + col] = (_Float16)fmaxf(a[r] + bb, 0.f);
    }
    __syncthreads();
    for(int ks = 0; ks < 8; ++ks){
      int k0 = khO*256 + ks*32;
      v16h av = load_a_frag(HID, 512, lane, k0);
      v16h bv = load_b_frag(fw2, 2048, lane, ntO*16, chunk*512 + k0);
      acc = wmma_f16(av, bv, acc);
    }
    __syncthreads();
  }
  {
    int col = ntO*16 + (lane & 15);
    int rb = (lane & 16) >> 1;
#pragma unroll
    for(int r = 0; r < 8; ++r) TMP2[khO*1024 + (rb + r)*64 + col] = acc[r];
  }
  __syncthreads();
  for(int i = tid; i < 1024; i += 256) TMP[i] = TMP2[i] + TMP2[1024 + i] + fb2[i & 63];
  __syncthreads();
  ln_add(S, TMP, ln2g, ln2b);
  __syncthreads();

  for(int i = tid; i < 1024; i += 256){
    float v = S[i] + (add_input ? X0[i] : 0.f);
    xout[(size_t)bn*1024 + i] = v;
  }
}

// ---------------- K3: mean over N + key padding mask ----------------
__global__ void pool_kernel(const float* __restrict__ xc, const float* __restrict__ pm,
                            float* __restrict__ xseq, float* __restrict__ kpm){
  int idx = blockIdx.x*blockDim.x + threadIdx.x;
  if(idx < B_*M_*D_){
    int b = idx / (M_*D_), rem = idx % (M_*D_), m = rem >> 6, d = rem & 63;
    float s = 0.f;
    for(int n = 0; n < N_; ++n) s += xc[((size_t)(b*N_ + n)*M_ + m)*64 + d];
    xseq[idx] = s*(1.f/(float)N_);
  }
  if(idx < B_*M_){
    int b = idx >> 4, m = idx & 15;
    float any = 0.f;
    for(int n = 0; n < N_; ++n) any += (pm[(b*N_ + n)*M_ + m] > 0.f) ? 1.f : 0.f;
    kpm[idx] = (any > 0.f) ? 0.f : 1.f;     // 1.0 => masked key
  }
}

// ---------------- K4: classifier head (one block per batch) ----------------
__global__ __launch_bounds__(128) void cls_kernel(
    const float* __restrict__ xseq, const float* __restrict__ kpm,
    const float* __restrict__ qe,
    const float* __restrict__ inw, const float* __restrict__ inb,
    const float* __restrict__ outw, const float* __restrict__ outb,
    const float* __restrict__ lw, const float* __restrict__ lb,
    float* __restrict__ out)
{
  __shared__ float xs[16*64], kk[16*64], vv[16*64], qq[5*64], att[5*64], zz[5*64];
  int b = blockIdx.x, tid = threadIdx.x;
  for(int i = tid; i < 1024; i += 128) xs[i] = xseq[(size_t)b*1024 + i];
  __syncthreads();
  for(int i = tid; i < 1024; i += 128){
    int m = i >> 6, d = i & 63;
    float k = inb[64 + d], v = inb[128 + d];
    for(int j = 0; j < 64; ++j){
      float xv = xs[m*64 + j];
      k += xv*inw[(64 + d)*64 + j];
      v += xv*inw[(128 + d)*64 + j];
    }
    kk[i] = k; vv[i] = v;
  }
  for(int i = tid; i < 320; i += 128){
    int q = i >> 6, d = i & 63;
    float a = inb[d];
    for(int j = 0; j < 64; ++j) a += qe[q*64 + j]*inw[d*64 + j];
    qq[i] = a;
  }
  __syncthreads();
  if(tid < 20){
    int qi = tid / 4, h = tid % 4;
    float sc[16]; float mx = -3.0e38f;
#pragma unroll
    for(int c = 0; c < 16; ++c){
      float s = 0.f;
#pragma unroll
      for(int d = 0; d < 16; ++d) s += qq[qi*64 + h*16 + d]*kk[c*64 + h*16 + d];
      s *= 0.25f;
      if(kpm[b*16 + c] != 0.f) s = -1.0e9f;
      sc[c] = s; mx = fmaxf(mx, s);
    }
    float sum = 0.f;
#pragma unroll
    for(int c = 0; c < 16; ++c){ sc[c] = __expf(sc[c] - mx); sum += sc[c]; }
    float inv = 1.f / sum;
#pragma unroll
    for(int d = 0; d < 16; ++d){
      float o = 0.f;
#pragma unroll
      for(int c = 0; c < 16; ++c) o += sc[c]*vv[c*64 + h*16 + d];
      att[qi*64 + h*16 + d] = o*inv;
    }
  }
  __syncthreads();
  for(int i = tid; i < 320; i += 128){
    int q = i >> 6, d = i & 63;
    float a = outb[d];
    for(int j = 0; j < 64; ++j) a += att[q*64 + j]*outw[d*64 + j];
    zz[i] = a;
  }
  __syncthreads();
  if(tid < 5){
    float a = lb[0];
    for(int d = 0; d < 64; ++d) a += zz[tid*64 + d]*lw[d];
    out[b*5 + tid] = a;
  }
}

// ---------------- host ----------------
extern "C" void kernel_launch(void* const* d_in, const int* in_sizes, int n_in,
                              void* d_out, int out_size, void* d_ws, size_t ws_size,
                              hipStream_t stream) {
  (void)in_sizes; (void)n_in; (void)out_size; (void)ws_size;
  const float* x      = (const float*)d_in[0];
  const int*   mask   = (const int*)  d_in[1];
  const float* w1     = (const float*)d_in[2];
  const float* b1     = (const float*)d_in[3];
  const float* w2     = (const float*)d_in[4];
  const float* b2     = (const float*)d_in[5];
  const float* w3     = (const float*)d_in[6];
  const float* b3     = (const float*)d_in[7];
  const float* tbias  = (const float*)d_in[8];
  const float* ga_qw  = (const float*)d_in[9];
  const float* ga_qb  = (const float*)d_in[10];
  const float* ga_kw  = (const float*)d_in[11];
  const float* ga_kb  = (const float*)d_in[12];
  const float* ga_vw  = (const float*)d_in[13];
  const float* ga_vb  = (const float*)d_in[14];
  const float* ga_ow  = (const float*)d_in[15];
  const float* ga_ob  = (const float*)d_in[16];
  const float* tf_inw = (const float*)d_in[17];
  const float* tf_inb = (const float*)d_in[18];
  const float* tf_ow  = (const float*)d_in[19];
  const float* tf_ob  = (const float*)d_in[20];
  const float* ln1g   = (const float*)d_in[21];
  const float* ln1b   = (const float*)d_in[22];
  const float* ln2g   = (const float*)d_in[23];
  const float* ln2b   = (const float*)d_in[24];
  const float* fw1    = (const float*)d_in[25];
  const float* fb1    = (const float*)d_in[26];
  const float* fw2    = (const float*)d_in[27];
  const float* fb2    = (const float*)d_in[28];
  const float* qe     = (const float*)d_in[29];
  const float* cinw   = (const float*)d_in[30];
  const float* cinb   = (const float*)d_in[31];
  const float* cow    = (const float*)d_in[32];
  const float* cob    = (const float*)d_in[33];
  const float* lw     = (const float*)d_in[34];
  const float* lb     = (const float*)d_in[35];
  float* out = (float*)d_out;

  // workspace carve (256B aligned)
  size_t off = 0;
  auto take = [&](size_t bytes){ size_t r = off; off = (off + bytes + 255) & ~(size_t)255; return r; };
  char* ws = (char*)d_ws;
  _Float16* w3h   = (_Float16*)(ws + take((size_t)W3N*W3K*2));
  _Float16* qw16  = (_Float16*)(ws + take((size_t)2*64*64*2));
  _Float16* kw16  = (_Float16*)(ws + take((size_t)2*64*64*2));
  _Float16* vw16  = (_Float16*)(ws + take((size_t)2*64*64*2));
  _Float16* ow16  = (_Float16*)(ws + take((size_t)2*64*64*2));
  _Float16* inw16 = (_Float16*)(ws + take((size_t)2*192*64*2));
  _Float16* ouw16 = (_Float16*)(ws + take((size_t)2*64*64*2));
  _Float16* fw116 = (_Float16*)(ws + take((size_t)2*2048*64*2));
  _Float16* fw216 = (_Float16*)(ws + take((size_t)2*64*2048*2));
  float* xc0   = (float*)(ws + take((size_t)P_*64*4));
  float* xc1   = (float*)(ws + take((size_t)P_*64*4));
  float* pmbuf = (float*)(ws + take((size_t)P_*4));
  float* xseq  = (float*)(ws + take((size_t)B_*M_*D_*4));
  float* kpmb  = (float*)(ws + take((size_t)B_*M_*4));

  auto cgrid = [](int n){ return dim3((n + 255)/256); };
  // weight conversions
  cvt_w3 <<<cgrid(W3N*W3K), 256, 0, stream>>>(w3, w3h);
  cvt_f16<<<cgrid(2*64*64), 256, 0, stream>>>(ga_qw, qw16, 2*64*64);
  cvt_f16<<<cgrid(2*64*64), 256, 0, stream>>>(ga_kw, kw16, 2*64*64);
  cvt_f16<<<cgrid(2*64*64), 256, 0, stream>>>(ga_vw, vw16, 2*64*64);
  cvt_f16<<<cgrid(2*64*64), 256, 0, stream>>>(ga_ow, ow16, 2*64*64);
  cvt_f16<<<cgrid(2*192*64), 256, 0, stream>>>(tf_inw, inw16, 2*192*64);
  cvt_f16<<<cgrid(2*64*64), 256, 0, stream>>>(tf_ow, ouw16, 2*64*64);
  cvt_f16<<<cgrid(2*2048*64), 256, 0, stream>>>(fw1, fw116, 2*2048*64);
  cvt_f16<<<cgrid(2*64*2048), 256, 0, stream>>>(fw2, fw216, 2*64*2048);

  // fused TTCN -> xc0 (= xp), pmbuf
  ttcn_kernel<<<dim3(P_), 256, 0, stream>>>(x, mask, w1, b1, w2, b2, w3h, b3, tbias, xc0, pmbuf);

  // transformer layers (ping-pong xc0 <-> xc1)
  for(int l = 0; l < 2; ++l){
    const float* in  = (l == 0) ? xc0 : xc1;
    float*       o   = (l == 0) ? xc1 : xc0;
    transformer_layer<<<dim3(B_*N_), 256, 0, stream>>>(
        in, o, pmbuf,
        qw16 + (size_t)l*4096, ga_qb + l*64,
        kw16 + (size_t)l*4096, ga_kb + l*64,
        vw16 + (size_t)l*4096, ga_vb + l*64,
        ow16 + (size_t)l*4096, ga_ob + l*64,
        inw16 + (size_t)l*192*64, tf_inb + l*192,
        ouw16 + (size_t)l*4096, tf_ob + l*64,
        ln1g + l*64, ln1b + l*64, ln2g + l*64, ln2b + l*64,
        fw116 + (size_t)l*2048*64, fb1 + l*2048,
        fw216 + (size_t)l*64*2048, fb2 + l*64,
        l > 0 ? 1 : 0);
  }

  // pool over N + key mask, then classifier head
  pool_kernel<<<cgrid(B_*M_*D_), 256, 0, stream>>>(xc0, pmbuf, xseq, kpmb);
  cls_kernel<<<dim3(B_), 128, 0, stream>>>(xseq, kpmb, qe, cinw, cinb, cow, cob, lw, lb, out);
}
